// ESTAM_69269232549926
// MI455X (gfx1250) — compile-verified, matrix operand order checked by reading, not run on hardware
//
#include <hip/hip_runtime.h>
#include <hip/hip_bf16.h>

typedef _Float16 v16h __attribute__((ext_vector_type(16)));
typedef _Float16 v8h  __attribute__((ext_vector_type(8)));
typedef float    v8f  __attribute__((ext_vector_type(8)));

// ---------- fast math helpers ----------
__device__ __forceinline__ float fexp(float x) {
#if __has_builtin(__builtin_amdgcn_exp2f)
    return __builtin_amdgcn_exp2f(x * 1.4426950408889634f);
#else
    return __expf(x);
#endif
}
__device__ __forceinline__ float frcp_(float x) {
#if __has_builtin(__builtin_amdgcn_rcpf)
    return __builtin_amdgcn_rcpf(x);
#else
    return 1.0f / x;
#endif
}
__device__ __forceinline__ float tanh_f(float x) {
#if __has_builtin(__builtin_amdgcn_tanhf)
    return __builtin_amdgcn_tanhf(x);            // v_tanh_f32: 1 trans op
#else
    float e = fexp(2.0f * x);
    return 1.0f - 2.0f * frcp_(e + 1.0f);
#endif
}
// sigmoid(x) = 0.5 + 0.5*tanh(x/2): 1 trans + 1 fma when v_tanh exists
__device__ __forceinline__ float sigm(float x) {
#if __has_builtin(__builtin_amdgcn_tanhf)
    return __builtin_fmaf(0.5f, __builtin_amdgcn_tanhf(0.5f * x), 0.5f);
#else
    return frcp_(1.0f + fexp(-x));
#endif
}
__device__ __forceinline__ void atomicMaxF(float* addr, float v) {
    if (v >= 0.0f) atomicMax((int*)addr, __float_as_int(v));
    else           atomicMin((unsigned int*)addr, __float_as_uint(v));
}

// Load a 16x32 (or 32x16) f16 WMMA fragment from a row-major [rows][64] f16 LDS array.
// Lane mapping per CDNA5 ISA 7.12.2: lanes 0-15 -> K {k0..k0+7, k0+16..k0+23},
// lanes 16-31 -> same +8. Caller passes k0 = kb*32 + (lane>>4)*8.
__device__ __forceinline__ v16h ldsFrag(const _Float16* base, int row, int k0) {
    const _Float16* p = base + row * 64 + k0;
    v8h lo = *(const v8h*)(p);
    v8h hi = *(const v8h*)(p + 16);
    v16h r;
#pragma unroll
    for (int i = 0; i < 8; ++i) { r[i] = lo[i]; r[8 + i] = hi[i]; }
    return r;
}

// =====================================================================
// Kernel 1: per-node LSTM (T=64, H=64) + GAT input projection (xh = h @ gat_w^T)
// One wave = 16 nodes. 8 waves / 256-thread block = 128 nodes per block.
// w_hh (f16) and gat_w (f16) live in LDS; h round-trips through a per-wave LDS tile
// to convert the WMMA C-layout back into the A-layout each timestep.
// launch_bounds(256, 2): require a register budget that admits >=2 waves/EU so a
// second block can be resident per WGP to hide the per-timestep barrier.
// =====================================================================
__global__ __launch_bounds__(256, 2) void estam_lstm_front(
    const float* __restrict__ x,      // [N, 64, 2]
    const float* __restrict__ w_ih,   // [256, 2]
    const float* __restrict__ w_hh,   // [256, 64]
    const float* __restrict__ b_ih,   // [256]
    const float* __restrict__ b_hh,   // [256]
    const float* __restrict__ gat_w,  // [128, 64]
    float* __restrict__ xh,           // [N, 128]
    int nNodes)
{
    __shared__ _Float16 whls[256 * 64];       // 32 KB: recurrent weights (B source)
    __shared__ _Float16 gwls[128 * 64];       // 16 KB: GAT weights
    __shared__ _Float16 hls_all[8 * 16 * 64]; // 16 KB: per-wave h tiles (f16)

    const int tid = threadIdx.x;
    for (int i = tid; i < 256 * 64; i += 256) whls[i] = (_Float16)w_hh[i];
    for (int i = tid; i < 128 * 64; i += 256) gwls[i] = (_Float16)gat_w[i];
    for (int i = tid; i < 8 * 16 * 64; i += 256) hls_all[i] = (_Float16)0.0f;
    __syncthreads();

    const int lane  = tid & 31;
    const int wave  = tid >> 5;
    const int idx16 = lane & 15;
    const int khalf = lane >> 4;
    const int k0b   = khalf * 8;

    // Clamp tail waves onto a valid tile (keeps EXEC all-ones for WMMA; duplicate
    // writes store identical values).
    int tile = blockIdx.x * 8 + wave;
    const int maxTile = nNodes / 16 - 1;
    if (tile > maxTile) tile = maxTile;
    const int nodeBase = tile * 16;

    _Float16* hls = hls_all + wave * (16 * 64);

    // per-lane input-projection constants: gate dim gd = 16*j + idx16
    float w0c[16], w1c[16], bbc[16];
#pragma unroll
    for (int j = 0; j < 16; ++j) {
        int gd = 16 * j + idx16;
        w0c[j] = w_ih[gd * 2 + 0];
        w1c[j] = w_ih[gd * 2 + 1];
        bbc[j] = b_ih[gd] + b_hh[gd];
    }

    v8f cst[4];                          // cell state, C-layout tiles over hidden dims
#pragma unroll
    for (int q = 0; q < 4; ++q)
#pragma unroll
        for (int r = 0; r < 8; ++r) cst[q][r] = 0.0f;

    for (int t = 0; t < 64; ++t) {
        // A fragments (h, 16x32 f16) from LDS
        v16h a0 = ldsFrag(hls, idx16, 0 + k0b);
        v16h a1 = ldsFrag(hls, idx16, 32 + k0b);

        // x values for this wave's 8 node-rows
        float xv0[8], xv1[8];
#pragma unroll
        for (int r = 0; r < 8; ++r) {
            int node = nodeBase + khalf * 8 + r;
            const float* xp = x + ((long long)node * 64 + t) * 2;
            xv0[r] = xp[0];
            xv1[r] = xp[1];
        }

        // Hidden-dim tile group q covers gate tiles {q, 4+q, 8+q, 12+q} = {i,f,g,o}.
#pragma unroll
        for (int q = 0; q < 4; ++q) {
            v8f acc[4];
#pragma unroll
            for (int g = 0; g < 4; ++g) {
                const int j = 4 * g + q;
#pragma unroll
                for (int r = 0; r < 8; ++r)
                    acc[g][r] = bbc[j] + w0c[j] * xv0[r] + w1c[j] * xv1[r];
            }
#pragma unroll
            for (int g = 0; g < 4; ++g) {
                const int j = 4 * g + q;
                v16h b0 = ldsFrag(whls, 16 * j + idx16, 0 + k0b);
                acc[g] = __builtin_amdgcn_wmma_f32_16x16x32_f16(
                    false, a0, false, b0, (short)0, acc[g], false, false);
                v16h b1 = ldsFrag(whls, 16 * j + idx16, 32 + k0b);
                acc[g] = __builtin_amdgcn_wmma_f32_16x16x32_f16(
                    false, a1, false, b1, (short)0, acc[g], false, false);
            }
            // activations + cell update for this hidden-dim group
#pragma unroll
            for (int r = 0; r < 8; ++r) {
                float iv = sigm(acc[0][r]);
                float fv = sigm(acc[1][r]);
                float gv = tanh_f(acc[2][r]);
                float ov = sigm(acc[3][r]);
                float cv = fv * cst[q][r] + iv * gv;
                cst[q][r] = cv;
                float hv = ov * tanh_f(cv);
                hls[(khalf * 8 + r) * 64 + q * 16 + idx16] = (_Float16)hv;
            }
        }
        __syncthreads();
    }

    // GAT projection: xh[16 nodes][128] = h @ gat_w^T (8 tiles x 2 K-blocks),
    // storing each tile immediately to keep register pressure low.
    {
        v16h a0 = ldsFrag(hls, idx16, 0 + k0b);
        v16h a1 = ldsFrag(hls, idx16, 32 + k0b);
#pragma unroll
        for (int j = 0; j < 8; ++j) {
            v8f z;
#pragma unroll
            for (int r = 0; r < 8; ++r) z[r] = 0.0f;
            v16h b0 = ldsFrag(gwls, 16 * j + idx16, 0 + k0b);
            z = __builtin_amdgcn_wmma_f32_16x16x32_f16(
                false, a0, false, b0, (short)0, z, false, false);
            v16h b1 = ldsFrag(gwls, 16 * j + idx16, 32 + k0b);
            z = __builtin_amdgcn_wmma_f32_16x16x32_f16(
                false, a1, false, b1, (short)0, z, false, false);
#pragma unroll
            for (int r = 0; r < 8; ++r) {
                int node = nodeBase + khalf * 8 + r;
                xh[(long long)node * 128 + j * 16 + idx16] = z[r];
            }
        }
    }
}

// =====================================================================
// Kernel 2: per-(node,head) attention coefficients a_src / a_dst
// =====================================================================
__global__ void estam_attn_coef(const float* __restrict__ xh,
                                const float* __restrict__ att_src,
                                const float* __restrict__ att_dst,
                                float* __restrict__ a_src,
                                float* __restrict__ a_dst, int nNodes)
{
    int i = blockIdx.x * blockDim.x + threadIdx.x; // node*2 + head
    if (i >= nNodes * 2) return;
    int h = i & 1, n = i >> 1;
    const float* v  = xh + (long long)n * 128 + h * 64;
    const float* as = att_src + h * 64;
    const float* ad = att_dst + h * 64;
    float s = 0.0f, d = 0.0f;
#pragma unroll 8
    for (int k = 0; k < 64; ++k) { float xv = v[k]; s += xv * as[k]; d += xv * ad[k]; }
    a_src[i] = s;
    a_dst[i] = d;
}

// =====================================================================
// Kernel 3: workspace init (segment max/sum + aggregation buffers)
// =====================================================================
__global__ void estam_init(float* __restrict__ mbuf, float* __restrict__ zbuf,
                           float* __restrict__ agg, int nNodes)
{
    long long i = (long long)blockIdx.x * blockDim.x + threadIdx.x;
    if (i < (long long)nNodes * 2) { mbuf[i] = -1e30f; zbuf[i] = 0.0f; }
    if (i < (long long)nNodes * 128) agg[i] = 0.0f;
}

// =====================================================================
// Kernel 4a: edge pass 1 — leaky-relu score + atomic float segment max
// (edges [0,E) from edge_index, [E, E+N) are self loops)
// =====================================================================
__global__ void estam_edge_max(const int* __restrict__ ei,
                               const float* __restrict__ a_src,
                               const float* __restrict__ a_dst,
                               float* __restrict__ ebuf,
                               float* __restrict__ mbuf,
                               int E, int nNodes)
{
    long long i = (long long)blockIdx.x * blockDim.x + threadIdx.x;
    long long Etot = (long long)E + nNodes;
    if (i >= Etot) return;
    int s, d;
    if (i < E) { s = ei[i]; d = ei[(long long)E + i]; }
    else       { s = d = (int)(i - E); }
#pragma unroll
    for (int h = 0; h < 2; ++h) {
        float e = a_src[s * 2 + h] + a_dst[d * 2 + h];
        e = e > 0.0f ? e : 0.2f * e;   // leaky relu
        ebuf[i * 2 + h] = e;
        atomicMaxF(&mbuf[d * 2 + h], e);
    }
}

// =====================================================================
// Kernel 4b: edge pass 2 — p = exp(e - m[dst]); segment sum z
// =====================================================================
__global__ void estam_edge_exp(const int* __restrict__ ei,
                               const float* __restrict__ mbuf,
                               float* __restrict__ ebuf,
                               float* __restrict__ zbuf,
                               int E, int nNodes)
{
    long long i = (long long)blockIdx.x * blockDim.x + threadIdx.x;
    long long Etot = (long long)E + nNodes;
    if (i >= Etot) return;
    int d;
    if (i < E) d = ei[(long long)E + i];
    else       d = (int)(i - E);
#pragma unroll
    for (int h = 0; h < 2; ++h) {
        float p = fexp(ebuf[i * 2 + h] - mbuf[d * 2 + h]);
        ebuf[i * 2 + h] = p;
        unsafeAtomicAdd(&zbuf[d * 2 + h], p);
    }
}

// =====================================================================
// Kernel 4c: edge pass 3 — alpha-weighted message scatter-add.
// One wave per edge; each lane owns 4 contiguous feature dims
// (lanes 0-15 -> head 0, lanes 16-31 -> head 1). agg (51MB) stays in L2.
// =====================================================================
__global__ __launch_bounds__(256) void estam_edge_aggr(
    const int* __restrict__ ei,
    const float* __restrict__ ebuf,
    const float* __restrict__ zbuf,
    const float* __restrict__ xh,
    float* __restrict__ agg,
    int E, int nNodes)
{
    const int lane = threadIdx.x & 31;
    long long edge = (long long)blockIdx.x * 8 + (threadIdx.x >> 5);
    long long Etot = (long long)E + nNodes;
    if (edge >= Etot) return;
    int s, d;
    if (edge < E) { s = ei[edge]; d = ei[(long long)E + edge]; }
    else          { s = d = (int)(edge - E); }
    const int h = lane >> 4;
    const float alpha = ebuf[edge * 2 + h] * frcp_(zbuf[d * 2 + h]);
    const float4 xv = *(const float4*)(xh + (long long)s * 128 + lane * 4);
    float* ap = agg + (long long)d * 128 + lane * 4;
    unsafeAtomicAdd(ap + 0, alpha * xv.x);
    unsafeAtomicAdd(ap + 1, alpha * xv.y);
    unsafeAtomicAdd(ap + 2, alpha * xv.z);
    unsafeAtomicAdd(ap + 3, alpha * xv.w);
}

// =====================================================================
// Kernel 5: head — mean over heads + bias + relu, 64->2 linear, sigmoid
// =====================================================================
__global__ void estam_head(const float* __restrict__ agg,
                           const float* __restrict__ gat_b,
                           const float* __restrict__ lin_w,
                           const float* __restrict__ lin_b,
                           float* __restrict__ out, int nNodes)
{
    int n = blockIdx.x * blockDim.x + threadIdx.x;
    if (n >= nNodes) return;
    const float* a = agg + (long long)n * 128;
    float y0 = lin_b[0], y1 = lin_b[1];
#pragma unroll 8
    for (int d = 0; d < 64; ++d) {
        float s = 0.5f * (a[d] + a[64 + d]) + gat_b[d];
        s = s > 0.0f ? s : 0.0f;
        y0 += s * lin_w[d];
        y1 += s * lin_w[64 + d];
    }
    out[n * 2 + 0] = sigm(y0);
    out[n * 2 + 1] = sigm(y1);
}

// =====================================================================
extern "C" void kernel_launch(void* const* d_in, const int* in_sizes, int n_in,
                              void* d_out, int out_size, void* d_ws, size_t ws_size,
                              hipStream_t stream)
{
    const float* x       = (const float*)d_in[0];
    const int*   ei      = (const int*)  d_in[1];
    const float* w_ih    = (const float*)d_in[2];
    const float* w_hh    = (const float*)d_in[3];
    const float* b_ih    = (const float*)d_in[4];
    const float* b_hh    = (const float*)d_in[5];
    const float* gat_w   = (const float*)d_in[6];
    const float* att_src = (const float*)d_in[7];
    const float* att_dst = (const float*)d_in[8];
    const float* gat_b   = (const float*)d_in[9];
    const float* lin_w   = (const float*)d_in[10];
    const float* lin_b   = (const float*)d_in[11];
    float* out = (float*)d_out;

    const int nNodes = in_sizes[0] / (64 * 2);
    const int E      = in_sizes[1] / 2;
    const long long Etot = (long long)E + nNodes;

    // workspace layout (floats)
    float* ws    = (float*)d_ws;
    float* xh    = ws;                              // N*128
    float* a_src = xh    + (long long)nNodes * 128; // N*2
    float* a_dst = a_src + (long long)nNodes * 2;   // N*2
    float* mbuf  = a_dst + (long long)nNodes * 2;   // N*2
    float* zbuf  = mbuf  + (long long)nNodes * 2;   // N*2
    float* agg   = zbuf  + (long long)nNodes * 2;   // N*128
    float* ebuf  = agg   + (long long)nNodes * 128; // (E+N)*2

    // 1) LSTM + GAT projection (WMMA)
    {
        int blocks = (nNodes + 127) / 128;
        estam_lstm_front<<<blocks, 256, 0, stream>>>(x, w_ih, w_hh, b_ih, b_hh,
                                                     gat_w, xh, nNodes);
    }
    // 2) init segment buffers
    {
        long long tot = (long long)nNodes * 128;
        int blocks = (int)((tot + 255) / 256);
        estam_init<<<blocks, 256, 0, stream>>>(mbuf, zbuf, agg, nNodes);
    }
    // 3) attention coefficients
    {
        int blocks = (nNodes * 2 + 255) / 256;
        estam_attn_coef<<<blocks, 256, 0, stream>>>(xh, att_src, att_dst,
                                                    a_src, a_dst, nNodes);
    }
    // 4) edge softmax (3 passes) + aggregation
    {
        int blocks = (int)((Etot + 255) / 256);
        estam_edge_max<<<blocks, 256, 0, stream>>>(ei, a_src, a_dst, ebuf, mbuf, E, nNodes);
        estam_edge_exp<<<blocks, 256, 0, stream>>>(ei, mbuf, ebuf, zbuf, E, nNodes);
        int ablocks = (int)((Etot + 7) / 8);
        estam_edge_aggr<<<ablocks, 256, 0, stream>>>(ei, ebuf, zbuf, xh, agg, E, nNodes);
    }
    // 5) output head
    {
        int blocks = (nNodes + 255) / 256;
        estam_head<<<blocks, 256, 0, stream>>>(agg, gat_b, lin_w, lin_b, out, nNodes);
    }
}
